// LEHD_66211215835191
// MI455X (gfx1250) — matrix operand order repaired and checked by reading, not compiled, not run on hardware
//
#include <hip/hip_runtime.h>
#include <hip/hip_bf16.h>
#include <cstdint>

// ---------------------------------------------------------------------------
// LEHD transformer forward on gfx1250 using V_WMMA_F32_16X16X32_BF16 with
// GLOBAL_LOAD_ASYNC_TO_LDS_B128 double-buffered weight staging.
// ---------------------------------------------------------------------------

typedef __attribute__((ext_vector_type(16))) __bf16 bf16x16;
typedef __attribute__((ext_vector_type(8)))  __bf16 bf16x8;
typedef __attribute__((ext_vector_type(8)))  float  f32x8;

#define WMMA_BF16(A, B, C) \
  __builtin_amdgcn_wmma_f32_16x16x32_bf16(false, (A), false, (B), (short)0, (C), false, false)

// Problem constants
#define NB   8
#define NS   512
#define ND   512
#define NH   8
#define NHD  64
#define NHID 2048
#define NL   6
#define NC   256
#define KIN  288      // 257 padded to multiple of 32
#define NM   4096     // NB*NS rows

// ---------------------------------------------------------------------------
// Fragment loaders.
// A-frag (16x32 bf16): lane half 0 -> K 0..15 contiguous, half 1 -> K 16..31
// B-frag (32x16 bf16): lane half 0 -> K 0..15 contiguous, half 1 -> K 16..31
// ---------------------------------------------------------------------------
__device__ __forceinline__ bf16x16 load_afrag(const __bf16* p) {
  union { bf16x16 v; bf16x8 h[2]; } u;
  u.h[0] = *(const bf16x8*)(p);
  u.h[1] = *(const bf16x8*)(p + 8);
  return u.v;
}
__device__ __forceinline__ bf16x16 load_bfrag(const __bf16* p) {
  return *(const bf16x16*)(p);
}

// Issue async global->LDS copy of one 64col x 32k bf16 B tile (4 KB).
// 128 threads: thread t covers col=t>>1, 32 bytes at seg=(t&1)*16 elements.
// INST_OFFSET applies to both global and LDS addresses (ISA 08, GV mode).
__device__ __forceinline__ void stage_b_async(const __bf16* Wt, int ldw, int n0, int k,
                                              __bf16* dst /* LDS, 64*32 elems */) {
  int t = threadIdx.x;            // 0..127
  int col = t >> 1, seg = (t & 1) * 16;
  const __bf16* g = Wt + (size_t)(n0 + col) * ldw + k + seg;
  unsigned lds = (unsigned)(uintptr_t)(dst + col * 32 + seg);
  asm volatile("global_load_async_to_lds_b128 %0, %1, off\n\t"
               "global_load_async_to_lds_b128 %0, %1, off offset:16"
               :: "v"(lds), "v"(g) : "memory");
}
__device__ __forceinline__ void wait_async0() {
  asm volatile("s_wait_asynccnt 0x0" ::: "memory");
}

// ---------------------------------------------------------------------------
// Weight convert + transpose: src fp32 (K, N) -> dst bf16 (N, Kpad), zero pad.
// ---------------------------------------------------------------------------
__global__ void convpose_kernel(const float* __restrict__ src, __bf16* __restrict__ dst,
                                int K, int N, int Kpad) {
  int i = blockIdx.x * blockDim.x + threadIdx.x;
  if (i >= N * Kpad) return;
  int n = i / Kpad, kk = i % Kpad;
  dst[i] = (kk < K) ? (__bf16)src[(size_t)kk * N + n] : (__bf16)0.0f;
}

// fp32 -> bf16 copy
__global__ void f32_to_bf16_kernel(const float* __restrict__ src, __bf16* __restrict__ dst, int n) {
  int i = blockIdx.x * blockDim.x + threadIdx.x;
  if (i < n) dst[i] = (__bf16)src[i];
}

// ---------------------------------------------------------------------------
// Build padded input features xin (NM, KIN) bf16:
//  cols 0..255 = embeddings[idx], col 256 = nd (0 for s==0), 257..287 = 0
// ---------------------------------------------------------------------------
__global__ void build_xin_kernel(const float* __restrict__ sol, const float* __restrict__ caps,
                                 const float* __restrict__ emb, __bf16* __restrict__ xin) {
  int row = blockIdx.x;                 // 0..4095
  int b = row >> 9, s = row & 511;
  int t = threadIdx.x;                  // 256 threads
  int idx = (int)sol[(size_t)row * 4 + 0];
  xin[(size_t)row * KIN + t] = (__bf16)emb[(size_t)idx * NC + t];
  if (t < 32) {
    float val = 0.0f;
    if (t == 0 && s != 0) {
      float rem = sol[(size_t)b * NS * 4 + 3];
      float dem = sol[(size_t)row * 4 + 2];
      val = 2.0f * (dem - rem) / caps[b];
    }
    xin[(size_t)row * KIN + NC + t] = (__bf16)val;
  }
}

// ---------------------------------------------------------------------------
// Generic bf16 WMMA GEMM: out = A(M,K) @ Wt(N,K)^T + bias
// block = 128 threads (4 waves); each wave computes a 32x64 tile; the shared
// 64x32 B tile is staged once per block through LDS via async loads,
// double-buffered so the fill overlaps the WMMA phase.
// grid = (N/64, M/128)
// modes: 0: outF = acc+bias          1: outF += acc+bias
//        2: outB = bf16(relu(acc+b)) 3: outB bf16 at (B,H,S,hd) (Q/K layout)
//        4: outB bf16 at (B,H,hd,S)  (V transposed layout)
// ---------------------------------------------------------------------------
__global__ __launch_bounds__(128) void gemm_wmma_kernel(
    const __bf16* __restrict__ A, int lda,
    const __bf16* __restrict__ Wt, int ldw,
    const float* __restrict__ bias,
    float* __restrict__ outF, __bf16* __restrict__ outB,
    int N, int K, int mode) {
  __shared__ __align__(32) __bf16 bstage[2][64 * 32];   // 8 KB double buffer
  const int lane = threadIdx.x & 31;
  const int wv   = threadIdx.x >> 5;
  const int l16  = lane & 15;
  const int half = lane >> 4;
  const int m0 = blockIdx.y * 128 + wv * 32;
  const int n0 = blockIdx.x * 64;

  f32x8 acc[2][4];
#pragma unroll
  for (int mt = 0; mt < 2; ++mt)
#pragma unroll
    for (int nt = 0; nt < 4; ++nt) acc[mt][nt] = (f32x8)0.0f;

  const __bf16* arow0 = A + (size_t)(m0 + l16) * lda + half * 16;
  const __bf16* arow1 = A + (size_t)(m0 + 16 + l16) * lda + half * 16;

  stage_b_async(Wt, ldw, n0, 0, bstage[0]);

  for (int k = 0; k < K; k += 32) {
    const int cur = (k >> 5) & 1;
    // wait for this iteration's B tile, make it visible to all 4 waves
    wait_async0();
    __syncthreads();
    // refill the other buffer (its readers all passed the barrier above);
    // this fill's latency hides under the WMMAs below.
    if (k + 32 < K) {
      stage_b_async(Wt, ldw, n0, k + 32, bstage[1 - cur]);
      __builtin_prefetch(arow0 + k + 32, 0, 1);
      __builtin_prefetch(arow1 + k + 32, 0, 1);
    }

    bf16x16 a0 = load_afrag(arow0 + k);
    bf16x16 a1 = load_afrag(arow1 + k);
    bf16x16 bf[4];
#pragma unroll
    for (int nt = 0; nt < 4; ++nt)
      bf[nt] = load_bfrag(&bstage[cur][(nt * 16 + l16) * 32 + half * 16]);
#pragma unroll
    for (int nt = 0; nt < 4; ++nt) {
      acc[0][nt] = WMMA_BF16(a0, bf[nt], acc[0][nt]);
      acc[1][nt] = WMMA_BF16(a1, bf[nt], acc[1][nt]);
    }
  }

#pragma unroll
  for (int mt = 0; mt < 2; ++mt) {
#pragma unroll
    for (int nt = 0; nt < 4; ++nt) {
#pragma unroll
      for (int i = 0; i < 8; ++i) {
        int row = m0 + mt * 16 + i + half * 8;   // C layout: M = vgpr + 8*half
        int col = n0 + nt * 16 + l16;            //           N = lane%16
        float val = acc[mt][nt][i] + bias[col];
        if (mode == 0) {
          outF[(size_t)row * N + col] = val;
        } else if (mode == 1) {
          outF[(size_t)row * N + col] += val;
        } else if (mode == 2) {
          outB[(size_t)row * N + col] = (__bf16)fmaxf(val, 0.0f);
        } else {
          int bb = row >> 9, ss = row & 511;
          int hh = col >> 6, dd = col & 63;
          if (mode == 3)
            outB[((size_t)(bb * NH + hh) * NS + ss) * NHD + dd] = (__bf16)val;
          else
            outB[((size_t)(bb * NH + hh) * NHD + dd) * NS + ss] = (__bf16)val;
        }
      }
    }
  }
}

// ---------------------------------------------------------------------------
// Recompute rows s==0 (src_W) and s==511 (dep_W), K=257, in fp32.
// grid = 16 blocks (b*2 + isdep), block = 256.
// ---------------------------------------------------------------------------
__global__ void fixup_rows_kernel(const float* __restrict__ sol, const float* __restrict__ caps,
                                  const float* __restrict__ emb,
                                  const float* __restrict__ srcW, const float* __restrict__ srcB,
                                  const float* __restrict__ depW, const float* __restrict__ depB,
                                  float* __restrict__ x) {
  int b = blockIdx.x >> 1;
  int isdep = blockIdx.x & 1;
  int s = isdep ? (NS - 1) : 0;
  int row = b * NS + s;
  const float* W  = isdep ? depW : srcW;   // (257, 512)
  const float* bi = isdep ? depB : srcB;
  __shared__ float xr[257];
  int idx = (int)sol[(size_t)row * 4 + 0];
  for (int t = threadIdx.x; t < NC; t += blockDim.x) xr[t] = emb[(size_t)idx * NC + t];
  if (threadIdx.x == 0) {
    float val = 0.0f;
    if (s != 0) {
      float rem = sol[(size_t)b * NS * 4 + 3];
      float dem = sol[(size_t)row * 4 + 2];
      val = 2.0f * (dem - rem) / caps[b];
    }
    xr[NC] = val;
  }
  __syncthreads();
  for (int c = threadIdx.x; c < ND; c += blockDim.x) {
    float a = bi[c];
    for (int kk = 0; kk < 257; ++kk) a += xr[kk] * W[(size_t)kk * ND + c];
    x[(size_t)row * ND + c] = a;
  }
}

// ---------------------------------------------------------------------------
// Flash attention per wave: one 16-row query tile of one (b,h).
// q,k: (B,H,S,hd) bf16 ; v: (B,H,hd,S) bf16 ; out ao: (B*S, D) bf16 concat.
// block = 256 (8 waves), grid = (B*H, S/128).
// Softmax shift s*log(n) is row-constant -> invariant, omitted.
// ---------------------------------------------------------------------------
__global__ __launch_bounds__(256) void attn_wmma_kernel(
    const __bf16* __restrict__ q, const __bf16* __restrict__ k,
    const __bf16* __restrict__ v, __bf16* __restrict__ ao) {
  __shared__ __align__(32) __bf16 pbuf[8][16 * 32];
  const int lane = threadIdx.x & 31;
  const int wv   = threadIdx.x >> 5;
  const int l16  = lane & 15;
  const int half = lane >> 4;
  const int bh = blockIdx.x;            // 0..63
  const int b = bh >> 3, h = bh & 7;
  const int s0 = (blockIdx.y * 8 + wv) * 16;

  const __bf16* qb = q + (size_t)bh * NS * NHD;
  const __bf16* kb = k + (size_t)bh * NS * NHD;
  const __bf16* vb = v + (size_t)bh * NHD * NS;

  bf16x16 qf[2];
#pragma unroll
  for (int kk = 0; kk < 2; ++kk)
    qf[kk] = load_afrag(qb + (size_t)(s0 + l16) * NHD + kk * 32 + half * 16);

  float mrow[8], lrow[8];
  f32x8 o[4];
#pragma unroll
  for (int i = 0; i < 8; ++i) { mrow[i] = -1e30f; lrow[i] = 0.0f; }
#pragma unroll
  for (int nt = 0; nt < 4; ++nt) o[nt] = (f32x8)0.0f;

  __bf16* pl = pbuf[wv];

  for (int kbk = 0; kbk < NS; kbk += 32) {
    f32x8 st[2];
    st[0] = (f32x8)0.0f; st[1] = (f32x8)0.0f;
#pragma unroll
    for (int jt = 0; jt < 2; ++jt)
#pragma unroll
      for (int kk = 0; kk < 2; ++kk) {
        bf16x16 bf = load_bfrag(kb + (size_t)(kbk + jt * 16 + l16) * NHD + kk * 32 + half * 16);
        st[jt] = WMMA_BF16(qf[kk], bf, st[jt]);
      }

    float pr0[8], pr1[8];
#pragma unroll
    for (int i = 0; i < 8; ++i) {
      float v0 = st[0][i] * 0.125f;   // 1/sqrt(hd)
      float v1 = st[1][i] * 0.125f;
      float mb = fmaxf(v0, v1);
      mb = fmaxf(mb, __shfl_xor(mb, 1));
      mb = fmaxf(mb, __shfl_xor(mb, 2));
      mb = fmaxf(mb, __shfl_xor(mb, 4));
      mb = fmaxf(mb, __shfl_xor(mb, 8));
      float mn   = fmaxf(mrow[i], mb);
      float corr = __expf(mrow[i] - mn);
      float p0 = __expf(v0 - mn);
      float p1 = __expf(v1 - mn);
      float ss = p0 + p1;
      ss += __shfl_xor(ss, 1);
      ss += __shfl_xor(ss, 2);
      ss += __shfl_xor(ss, 4);
      ss += __shfl_xor(ss, 8);
      lrow[i] = lrow[i] * corr + ss;
      mrow[i] = mn;
#pragma unroll
      for (int nt = 0; nt < 4; ++nt) o[nt][i] *= corr;
      pr0[i] = p0; pr1[i] = p1;
    }

    // Transpose P (16x32) to A-fragment layout via private LDS tile.
#pragma unroll
    for (int i = 0; i < 8; ++i) {
      pl[(i + 8 * half) * 32 + l16]      = (__bf16)pr0[i];
      pl[(i + 8 * half) * 32 + 16 + l16] = (__bf16)pr1[i];
    }
    bf16x16 pf = load_afrag(pl + l16 * 32 + half * 16);   // same wave: DScnt in-order

#pragma unroll
    for (int nt = 0; nt < 4; ++nt) {
      bf16x16 vf = load_bfrag(vb + (size_t)(nt * 16 + l16) * NS + kbk + half * 16);
      o[nt] = WMMA_BF16(pf, vf, o[nt]);
    }
  }

#pragma unroll
  for (int nt = 0; nt < 4; ++nt)
#pragma unroll
    for (int i = 0; i < 8; ++i) {
      int srow = s0 + i + 8 * half;
      float val = o[nt][i] / lrow[i];
      ao[((size_t)(b * NS + srow)) * ND + h * NHD + nt * 16 + l16] = (__bf16)val;
    }
}

// ---------------------------------------------------------------------------
// feas / ptr heads (VALU; tiny FLOP counts).
// ---------------------------------------------------------------------------
__global__ void feas_kernel(const float* __restrict__ x, const float* __restrict__ fW,
                            const float* __restrict__ fb, float* __restrict__ out) {
  int t = blockIdx.x * blockDim.x + threadIdx.x;
  if (t >= NB * (NS - 2)) return;
  int b = t / (NS - 2), j = t % (NS - 2);
  const float* xr = x + ((size_t)b * NS + 1 + j) * ND;
  float a = fb[0];
  for (int kk = 0; kk < ND; ++kk) a += xr[kk] * fW[kk];
  out[NB * 2 * (NS - 2) + b * (NS - 2) + j] = a;
}

__global__ void ptr_kernel(const float* __restrict__ x, const float* __restrict__ pW,
                           const float* __restrict__ pb, float* __restrict__ out) {
  int t = blockIdx.x * blockDim.x + threadIdx.x;
  if (t >= NB * (NS - 2)) return;
  int b = t / (NS - 2), j = t % (NS - 2);
  const float* xr = x + ((size_t)b * NS + 1 + j) * ND;
  float a0 = pb[0], a1 = pb[1];
  for (int kk = 0; kk < ND; ++kk) {
    float xv = xr[kk];
    a0 += xv * pW[kk * 2];
    a1 += xv * pW[kk * 2 + 1];
  }
  out[b * 2 * (NS - 2) + j] = a0;
  out[b * 2 * (NS - 2) + (NS - 2) + j] = a1;
}

// ---------------------------------------------------------------------------
// Host orchestration
// ---------------------------------------------------------------------------
extern "C" void kernel_launch(void* const* d_in, const int* in_sizes, int n_in,
                              void* d_out, int out_size, void* d_ws, size_t ws_size,
                              hipStream_t stream) {
  const float* sol   = (const float*)d_in[0];
  const float* caps  = (const float*)d_in[1];
  const float* emb   = (const float*)d_in[2];
  const float* srcW  = (const float*)d_in[3];
  const float* srcB  = (const float*)d_in[4];
  const float* candW = (const float*)d_in[5];
  const float* candB = (const float*)d_in[6];
  const float* depW  = (const float*)d_in[7];
  const float* depB  = (const float*)d_in[8];
  const float* feasW = (const float*)d_in[9];
  const float* feasB = (const float*)d_in[10];
  const float* Wq = (const float*)d_in[11];
  const float* bq = (const float*)d_in[12];
  const float* Wk = (const float*)d_in[13];
  const float* bk = (const float*)d_in[14];
  const float* Wv = (const float*)d_in[15];
  const float* bv = (const float*)d_in[16];
  const float* Wo = (const float*)d_in[17];
  const float* bo = (const float*)d_in[18];
  const float* W1 = (const float*)d_in[20];
  const float* b1 = (const float*)d_in[21];
  const float* W2 = (const float*)d_in[22];
  const float* b2 = (const float*)d_in[23];
  const float* pW = (const float*)d_in[24];
  const float* pb = (const float*)d_in[25];
  float* out = (float*)d_out;

  // Workspace carve-out (256B aligned)
  char* p = (char*)d_ws;
  auto alloc = [&](size_t bytes) -> void* {
    void* r = (void*)p;
    p += (bytes + 255) & ~(size_t)255;
    return r;
  };
  __bf16* xin_bf = (__bf16*)alloc((size_t)NM * KIN * 2);
  float*  x_f32  = (float*) alloc((size_t)NM * ND * 4);
  __bf16* x_bf   = (__bf16*)alloc((size_t)NM * ND * 2);
  __bf16* q_bf   = (__bf16*)alloc((size_t)NB * NH * NS * NHD * 2);
  __bf16* k_bf   = (__bf16*)alloc((size_t)NB * NH * NS * NHD * 2);
  __bf16* v_bf   = (__bf16*)alloc((size_t)NB * NH * NHD * NS * 2);
  __bf16* ao_bf  = (__bf16*)alloc((size_t)NM * ND * 2);
  __bf16* h_bf   = (__bf16*)alloc((size_t)NM * NHID * 2);
  __bf16* wc_bf  = (__bf16*)alloc((size_t)ND * KIN * 2);
  __bf16* wq_bf  = (__bf16*)alloc((size_t)NL * ND * ND * 2);
  __bf16* wk_bf  = (__bf16*)alloc((size_t)NL * ND * ND * 2);
  __bf16* wv_bf  = (__bf16*)alloc((size_t)NL * ND * ND * 2);
  __bf16* wo_bf  = (__bf16*)alloc((size_t)NL * ND * ND * 2);
  __bf16* w1_bf  = (__bf16*)alloc((size_t)NL * NHID * ND * 2);
  __bf16* w2_bf  = (__bf16*)alloc((size_t)NL * ND * NHID * 2);

  auto convpose = [&](const float* src, __bf16* dst, int K, int N, int Kpad) {
    int total = N * Kpad;
    convpose_kernel<<<(total + 255) / 256, 256, 0, stream>>>(src, dst, K, N, Kpad);
  };
  auto gemm = [&](const __bf16* A, int lda, const __bf16* Wt, int ldw, const float* bias,
                  float* oF, __bf16* oB, int N, int K, int mode) {
    dim3 grid(N / 64, NM / 128);
    gemm_wmma_kernel<<<grid, 128, 0, stream>>>(A, lda, Wt, ldw, bias, oF, oB, N, K, mode);
  };

  // 1) inputs
  build_xin_kernel<<<NM, 256, 0, stream>>>(sol, caps, emb, xin_bf);

  // 2) weight conversions (transposed bf16)
  convpose(candW, wc_bf, 257, ND, KIN);
  for (int i = 0; i < NL; ++i) {
    convpose(Wq + (size_t)i * ND * ND, wq_bf + (size_t)i * ND * ND, ND, ND, ND);
    convpose(Wk + (size_t)i * ND * ND, wk_bf + (size_t)i * ND * ND, ND, ND, ND);
    convpose(Wv + (size_t)i * ND * ND, wv_bf + (size_t)i * ND * ND, ND, ND, ND);
    convpose(Wo + (size_t)i * ND * ND, wo_bf + (size_t)i * ND * ND, ND, ND, ND);
    convpose(W1 + (size_t)i * ND * NHID, w1_bf + (size_t)i * NHID * ND, ND, NHID, ND);
    convpose(W2 + (size_t)i * NHID * ND, w2_bf + (size_t)i * ND * NHID, NHID, ND, NHID);
  }

  // 3) input projection (cand weights everywhere), then fix src/dep rows
  gemm(xin_bf, KIN, wc_bf, KIN, candB, x_f32, nullptr, ND, KIN, 0);
  fixup_rows_kernel<<<16, 256, 0, stream>>>(sol, caps, emb, srcW, srcB, depW, depB, x_f32);

  int nX = NM * ND;
  auto mha = [&](int i) {
    f32_to_bf16_kernel<<<(nX + 255) / 256, 256, 0, stream>>>(x_f32, x_bf, nX);
    gemm(x_bf, ND, wq_bf + (size_t)i * ND * ND, ND, bq + (size_t)i * ND, nullptr, q_bf, ND, ND, 3);
    gemm(x_bf, ND, wk_bf + (size_t)i * ND * ND, ND, bk + (size_t)i * ND, nullptr, k_bf, ND, ND, 3);
    gemm(x_bf, ND, wv_bf + (size_t)i * ND * ND, ND, bv + (size_t)i * ND, nullptr, v_bf, ND, ND, 4);
    attn_wmma_kernel<<<dim3(NB * NH, NS / 128), 256, 0, stream>>>(q_bf, k_bf, v_bf, ao_bf);
    gemm(ao_bf, ND, wo_bf + (size_t)i * ND * ND, ND, bo + (size_t)i * ND, x_f32, nullptr, ND, ND, 1);
  };
  auto ffn = [&](int i) {
    f32_to_bf16_kernel<<<(nX + 255) / 256, 256, 0, stream>>>(x_f32, x_bf, nX);
    gemm(x_bf, ND, w1_bf + (size_t)i * NHID * ND, ND, b1 + (size_t)i * NHID, nullptr, h_bf, NHID, ND, 2);
    gemm(h_bf, NHID, w2_bf + (size_t)i * ND * NHID, NHID, b2 + (size_t)i * ND, x_f32, nullptr, ND, NHID, 1);
  };

  // 4) layer 0 MHA, feasibility head, then remaining layers
  mha(0);
  feas_kernel<<<(NB * (NS - 2) + 255) / 256, 256, 0, stream>>>(x_f32, feasW, feasB, out);
  for (int i = 0; i < NL; ++i) {
    if (i > 0) mha(i);
    ffn(i);
  }

  // 5) pointer head -> first NB*1020 floats of d_out
  ptr_kernel<<<(NB * (NS - 2) + 255) / 256, 256, 0, stream>>>(x_f32, pW, pb, out);

  (void)in_sizes; (void)n_in; (void)out_size; (void)ws_size;
}
